// HawkMixer_53386443489945
// MI455X (gfx1250) — compile-verified, matrix operand order checked by reading, not run on hardware
//
#include <hip/hip_runtime.h>
#include <hip/hip_bf16.h>
#include <math.h>

// ---------------------------------------------------------------------------
// Hawk mixer forward for MI455X (gfx1250), bf16 WMMA matrix path with
// double-buffered async global->LDS staging (GLOBAL_LOAD_ASYNC_TO_LDS_B128,
// ASYNCcnt software pipeline).
//   B=4, S=4096, D=1024, H=1536, K=4, C_CONST=8
// ---------------------------------------------------------------------------

typedef __attribute__((ext_vector_type(16))) __bf16 v16bf;
typedef __attribute__((ext_vector_type(8)))  float  v8f;

// exact param types for __builtin_amdgcn_global_load_async_to_lds_b128
// (from compiler diagnostic: gcc-style int4 vector, as1 src / as3 dst)
typedef int v4i_gcc __attribute__((vector_size(16)));
typedef __attribute__((address_space(1))) v4i_gcc global_v4i;
typedef __attribute__((address_space(3))) v4i_gcc lds_v4i;

#define DIMD 1024
#define HID  1536
#define BATCH 4
#define SEQ  4096
#define MTOT (BATCH * SEQ)   // 16384

#define BM 128
#define BN 128
#define BK 32

#if defined(__gfx1250__) && __has_builtin(__builtin_amdgcn_global_load_async_to_lds_b128)
#define HAS_ASYNC_LDS 1
#else
#define HAS_ASYNC_LDS 0
#endif

__device__ __forceinline__ void async_ld128(const void* g, void* l) {
#if HAS_ASYNC_LDS
  __builtin_amdgcn_global_load_async_to_lds_b128(
      (global_v4i*)(uintptr_t)g,
      (lds_v4i*)(unsigned)(uintptr_t)l,
      0, 0);
#else
  *(uint4*)l = *(const uint4*)g;
#endif
}

template <int N>
__device__ __forceinline__ void wait_async_lds() {
#if HAS_ASYNC_LDS
#if __has_builtin(__builtin_amdgcn_s_wait_asynccnt)
  __builtin_amdgcn_s_wait_asynccnt(N);
#else
  asm volatile("s_wait_asynccnt %0" ::"i"(N) : "memory");
#endif
#endif
}

// ---------------- f32 -> bf16 convert ----------------
__global__ void cvt_bf16_kernel(const float* __restrict__ in,
                                __bf16* __restrict__ out, size_t n) {
  size_t i = (size_t)blockIdx.x * blockDim.x + threadIdx.x;
  if (i < n) out[i] = (__bf16)in[i];
}

// ---------------- tiled bf16 WMMA GEMM: C[M,N] = A[M,K] * B[N,K]^T (+bias[N])
// block tile 128x128, 8 waves, each wave 64x32 (4x2 WMMA tiles of 16x16x32).
// Double-buffered LDS; async loads for tile t+1 overlap WMMA on tile t.
__global__ __launch_bounds__(256)
void gemm_bf16_kernel(const __bf16* __restrict__ A, const __bf16* __restrict__ B,
                      const float* __restrict__ bias, float* __restrict__ C,
                      int M, int N, int Kdim) {
  __shared__ __align__(16) __bf16 As[2][BM][BK];
  __shared__ __align__(16) __bf16 Bs[2][BN][BK];

  const int tid  = threadIdx.x;
  const int wid  = tid >> 5;
  const int lane = tid & 31;
  const int bm = blockIdx.y * BM;
  const int bn = blockIdx.x * BN;
  const int wm = (wid >> 2) * 64;   // 0,64
  const int wn = (wid & 3) * 32;    // 0,32,64,96

  v8f acc[4][2] = {};

  // cooperative global->LDS fill: each thread moves 4 x 16B per K-step
  const int lrow = tid >> 2;        // 0..63
  const int lcol = (tid & 3) * 8;   // 0,8,16,24

  const int frow = lane & 15;
  const int kh   = (lane >> 4) * 8; // K-half select per ISA 16-bit A/B layout

  union FU { v16bf v; uint4 q[2]; };

  auto issue_tile = [&](int k0, int buf) {
    async_ld128(A + (size_t)(bm + lrow)      * Kdim + k0 + lcol, &As[buf][lrow][lcol]);
    async_ld128(A + (size_t)(bm + lrow + 64) * Kdim + k0 + lcol, &As[buf][lrow + 64][lcol]);
    async_ld128(B + (size_t)(bn + lrow)      * Kdim + k0 + lcol, &Bs[buf][lrow][lcol]);
    async_ld128(B + (size_t)(bn + lrow + 64) * Kdim + k0 + lcol, &Bs[buf][lrow + 64][lcol]);
  };

  const int nk = Kdim / BK;
  issue_tile(0, 0);                 // prologue: tile 0 in flight

  for (int t = 0; t < nk; ++t) {
    const int k0  = t * BK;
    const int cur = t & 1;
    if (t + 1 < nk) {
      issue_tile(k0 + BK, (t + 1) & 1);   // next tile streams while we compute
      if (k0 + 2 * BK < Kdim) {           // L2 prefetch two slabs ahead
        __builtin_prefetch(A + (size_t)(bm + lrow) * Kdim + k0 + 2 * BK + lcol, 0, 3);
        __builtin_prefetch(B + (size_t)(bn + lrow) * Kdim + k0 + 2 * BK + lcol, 0, 3);
      }
      wait_async_lds<4>();   // oldest 4 (tile t) done; tile t+1 still in flight
    } else {
      wait_async_lds<0>();
    }
    __syncthreads();         // all waves' tile-t data visible in LDS

    v16bf afrag[4], bfrag[2];
#pragma unroll
    for (int i = 0; i < 4; ++i) {
      FU f;
      f.q[0] = *(const uint4*)(&As[cur][wm + 16 * i + frow][kh]);
      f.q[1] = *(const uint4*)(&As[cur][wm + 16 * i + frow][kh + 16]);
      afrag[i] = f.v;
    }
#pragma unroll
    for (int j = 0; j < 2; ++j) {
      FU f;
      f.q[0] = *(const uint4*)(&Bs[cur][wn + 16 * j + frow][kh]);
      f.q[1] = *(const uint4*)(&Bs[cur][wn + 16 * j + frow][kh + 16]);
      bfrag[j] = f.v;
    }
#pragma unroll
    for (int i = 0; i < 4; ++i)
#pragma unroll
      for (int j = 0; j < 2; ++j)
        acc[i][j] = __builtin_amdgcn_wmma_f32_16x16x32_bf16(
            false, afrag[i], false, bfrag[j], (short)0, acc[i][j], false, false);

    __syncthreads();         // buffer `cur` safe to overwrite at t+2's issue
  }

  // epilogue: D layout -> row r + 8*(lane>>4), col lane&15
  const int crow0 = (lane >> 4) * 8;
  const int ccol  = lane & 15;
#pragma unroll
  for (int i = 0; i < 4; ++i)
#pragma unroll
    for (int j = 0; j < 2; ++j) {
      int gr0 = bm + wm + 16 * i + crow0;
      int gc  = bn + wn + 16 * j + ccol;
      float bv = bias ? bias[gc] : 0.0f;
#pragma unroll
      for (int r = 0; r < 8; ++r)
        C[(size_t)(gr0 + r) * N + gc] = acc[i][j][r] + bv;
    }
}

// ---------------- causal depthwise conv1d (K=4) + bias, emit bf16 ----------
__global__ void conv_kernel(const float* __restrict__ u,
                            const float* __restrict__ w,   // [H,1,4]
                            const float* __restrict__ b,   // [H]
                            __bf16* __restrict__ yb) {
  size_t idx = (size_t)blockIdx.x * blockDim.x + threadIdx.x;
  const size_t total = (size_t)MTOT * HID;
  if (idx >= total) return;
  int h = (int)(idx % HID);
  size_t bs = idx / HID;          // b*S + s
  int s = (int)(bs % SEQ);
  float acc = b[h];
#pragma unroll
  for (int k = 0; k < 4; ++k) {
    int sp = s - 3 + k;
    if (sp >= 0)
      acc += w[h * 4 + k] * u[(bs + (size_t)(sp - s)) * HID + h];
  }
  yb[idx] = (__bf16)acc;
}

// ---------------- sp[h] = -8 * softplus(lam[h]) ----------------------------
__global__ void sp_table_kernel(const float* __restrict__ lam,
                                float* __restrict__ sp) {
  int h = blockIdx.x * blockDim.x + threadIdx.x;
  if (h < HID) sp[h] = -8.0f * log1pf(expf(lam[h]));
}

// ---------------- alpha/xbeta elementwise (in place over igp/rgp) ----------
__global__ void ew_kernel(float* __restrict__ igp, float* __restrict__ rgp,
                          const __bf16* __restrict__ yb,
                          const float* __restrict__ sp) {
  size_t idx = (size_t)blockIdx.x * blockDim.x + threadIdx.x;
  const size_t total = (size_t)MTOT * HID;
  if (idx >= total) return;
  int h = (int)(idx % HID);
  float ig = 1.0f / (1.0f + expf(-igp[idx]));
  float rg = 1.0f / (1.0f + expf(-rgp[idx]));
  float a  = expf(sp[h] * rg);
  float beta = sqrtf(1.0f - a * a + 1e-6f);
  igp[idx] = a;                          // alpha
  rgp[idx] = beta * ig * (float)yb[idx]; // xbeta
}

// ---------------- diagonal linear recurrence: 6144 serial chains -----------
__global__ void scan_kernel(const float* __restrict__ alpha,
                            const float* __restrict__ xbeta,
                            float* __restrict__ hout) {
  int id = blockIdx.x * blockDim.x + threadIdx.x;   // b*H + h
  if (id >= BATCH * HID) return;
  int b = id / HID;
  int h = id % HID;
  float acc = 0.0f;
  size_t base = (size_t)b * SEQ * HID + h;
  for (int t = 0; t < SEQ; ++t) {
    size_t idx = base + (size_t)t * HID;
    acc = alpha[idx] * acc + xbeta[idx];
    hout[idx] = acc;
  }
}

// ---------------- z = bf16(gelu(gate) * h) ---------------------------------
__global__ void z_kernel(const float* __restrict__ gate,
                         const float* __restrict__ hbuf,
                         __bf16* __restrict__ zb) {
  size_t idx = (size_t)blockIdx.x * blockDim.x + threadIdx.x;
  const size_t total = (size_t)MTOT * HID;
  if (idx >= total) return;
  float g = gate[idx];
  float ge = 0.5f * g * (1.0f + erff(g * 0.70710678118654752f));
  zb[idx] = (__bf16)(ge * hbuf[idx]);
}

// ---------------------------------------------------------------------------
extern "C" void kernel_launch(void* const* d_in, const int* in_sizes, int n_in,
                              void* d_out, int out_size, void* d_ws, size_t ws_size,
                              hipStream_t stream) {
  const float* x      = (const float*)d_in[0];
  const float* Wg     = (const float*)d_in[1];
  const float* Wi     = (const float*)d_in[2];
  const float* conv_w = (const float*)d_in[3];
  const float* conv_b = (const float*)d_in[4];
  const float* lru_Wi = (const float*)d_in[5];
  const float* lru_bi = (const float*)d_in[6];
  const float* lru_Wg = (const float*)d_in[7];
  const float* lru_bg = (const float*)d_in[8];
  const float* lam    = (const float*)d_in[9];
  const float* Wo     = (const float*)d_in[10];
  float* out = (float*)d_out;

  // ---- workspace bump allocator ----
  char* ws = (char*)d_ws;
  size_t off = 0;
  auto alloc = [&](size_t bytes) -> void* {
    void* p = ws + off;
    off += (bytes + 255) & ~(size_t)255;
    return p;
  };
  const size_t MH = (size_t)MTOT * HID;
  const size_t MD = (size_t)MTOT * DIMD;

  __bf16* xb   = (__bf16*)alloc(MD * 2);
  __bf16* wgb  = (__bf16*)alloc((size_t)HID * DIMD * 2);
  __bf16* wib  = (__bf16*)alloc((size_t)HID * DIMD * 2);
  __bf16* wlib = (__bf16*)alloc((size_t)HID * HID * 2);
  __bf16* wlgb = (__bf16*)alloc((size_t)HID * HID * 2);
  __bf16* wob  = (__bf16*)alloc((size_t)DIMD * HID * 2);
  float*  gate = (float*)alloc(MH * 4);
  float*  ubuf = (float*)alloc(MH * 4);   // later reused as h
  __bf16* yb   = (__bf16*)alloc(MH * 2);
  float*  igp  = (float*)alloc(MH * 4);   // -> alpha
  float*  rgp  = (float*)alloc(MH * 4);   // -> xbeta
  __bf16* zb   = (__bf16*)alloc(MH * 2);
  float*  sp   = (float*)alloc((size_t)HID * 4);

  auto grid1d = [](size_t n) { return dim3((unsigned)((n + 255) / 256)); };

  // ---- convert to bf16 ----
  cvt_bf16_kernel<<<grid1d(MD), 256, 0, stream>>>(x, xb, MD);
  cvt_bf16_kernel<<<grid1d((size_t)HID * DIMD), 256, 0, stream>>>(Wg, wgb, (size_t)HID * DIMD);
  cvt_bf16_kernel<<<grid1d((size_t)HID * DIMD), 256, 0, stream>>>(Wi, wib, (size_t)HID * DIMD);
  cvt_bf16_kernel<<<grid1d((size_t)HID * HID), 256, 0, stream>>>(lru_Wi, wlib, (size_t)HID * HID);
  cvt_bf16_kernel<<<grid1d((size_t)HID * HID), 256, 0, stream>>>(lru_Wg, wlgb, (size_t)HID * HID);
  cvt_bf16_kernel<<<grid1d((size_t)DIMD * HID), 256, 0, stream>>>(Wo, wob, (size_t)DIMD * HID);
  sp_table_kernel<<<dim3((HID + 255) / 256), 256, 0, stream>>>(lam, sp);

  // ---- GEMM1/2: gate, u ----
  {
    dim3 g(HID / BN, MTOT / BM);
    gemm_bf16_kernel<<<g, 256, 0, stream>>>(xb, wgb, nullptr, gate, MTOT, HID, DIMD);
    gemm_bf16_kernel<<<g, 256, 0, stream>>>(xb, wib, nullptr, ubuf, MTOT, HID, DIMD);
  }

  // ---- conv -> y (bf16) ----
  conv_kernel<<<grid1d(MH), 256, 0, stream>>>(ubuf, conv_w, conv_b, yb);

  // ---- GEMM3/4: ig/rg pre-activations (+bias) ----
  {
    dim3 g(HID / BN, MTOT / BM);
    gemm_bf16_kernel<<<g, 256, 0, stream>>>(yb, wlib, lru_bi, igp, MTOT, HID, HID);
    gemm_bf16_kernel<<<g, 256, 0, stream>>>(yb, wlgb, lru_bg, rgp, MTOT, HID, HID);
  }

  // ---- alpha/xbeta ----
  ew_kernel<<<grid1d(MH), 256, 0, stream>>>(igp, rgp, yb, sp);

  // ---- recurrence (h into ubuf) ----
  scan_kernel<<<dim3((BATCH * HID + 255) / 256), 256, 0, stream>>>(igp, rgp, ubuf);

  // ---- z = gelu(gate) * h ----
  z_kernel<<<grid1d(MH), 256, 0, stream>>>(gate, ubuf, zb);

  // ---- GEMM5: out ----
  {
    dim3 g(DIMD / BN, MTOT / BM);
    gemm_bf16_kernel<<<g, 256, 0, stream>>>(zb, wob, nullptr, out, MTOT, DIMD, HID);
  }
  (void)in_sizes; (void)n_in; (void)out_size; (void)ws_size;
}
